// N3Aggregation2D_21912923144705
// MI455X (gfx1250) — compile-verified, matrix operand order checked by reading, not run on hardware
//
#include <hip/hip_runtime.h>
#include <hip/hip_bf16.h>
#include <math.h>

typedef __attribute__((ext_vector_type(16))) _Float16 v16h;
typedef __attribute__((ext_vector_type(8)))  float    v8f;
typedef unsigned int v4u __attribute__((ext_vector_type(4)));
typedef int          v8i __attribute__((ext_vector_type(8)));
typedef int          v4i __attribute__((ext_vector_type(4)));

constexpr int HD   = 64;        // data H=W
constexpr int HP   = 66;        // padded H=W
constexpr int NQ   = HP * HP;   // 4356 queries
constexpr int NE   = 8;         // embed channels
constexpr int NC   = 8;         // x channels
constexpr int NK   = 7;         // softmax rounds
constexpr int PS   = 10;        // patch size
constexpr int ADJ  = 5;
constexpr int WOFF = 14;        // 29//2
constexpr int NW   = 841;       // 29*29 offsets
constexpr int KSEL = 224;       // top-k
constexpr int XP   = 75;        // xpad spatial dim (66+9)
constexpr int XSTR = 76;        // row stride (conflict padding)
constexpr int BUFN = NK * NC * XP * XP;     // overlap-add buffer 56*75*75
constexpr int XPH  = NC * XP * XSTR;        // global f16 xpad elements (45600*? no: 8*75*76)
constexpr int XPASS = 4 * XP * XSTR;        // f16 elements per channel pass (22800)

// ---------------------------------------------------------------------------
// Kernel A: box-summed squared norms of ye/xe and box-averaged log_temp.
// ---------------------------------------------------------------------------
__global__ __launch_bounds__(256) void kprep(const float* __restrict__ xe,
                                             const float* __restrict__ ye,
                                             const float* __restrict__ ltin,
                                             float* __restrict__ n2q,
                                             float* __restrict__ n2db,
                                             float* __restrict__ ltv) {
  int i = blockIdx.x * blockDim.x + threadIdx.x;
  if (i >= NQ) return;
  int y = i / HP, x = i % HP;
  float sq = 0.f, sdb = 0.f, sl = 0.f;
  for (int oy = -ADJ; oy < PS - ADJ; ++oy) {
    int yy = y + oy;
    if (yy < 1 || yy > HD) continue;
    for (int ox = -ADJ; ox < PS - ADJ; ++ox) {
      int xx = x + ox;
      if (xx < 1 || xx > HD) continue;
      int base = (yy - 1) * HD + (xx - 1);
#pragma unroll
      for (int e = 0; e < NE; ++e) {
        float a = ye[e * HD * HD + base]; sq  += a * a;
        float b = xe[e * HD * HD + base]; sdb += b * b;
      }
      sl += ltin[base];
    }
  }
  n2q[i] = sq; n2db[i] = sdb; ltv[i] = sl * (1.0f / (PS * PS));
}

// ---------------------------------------------------------------------------
// Kernel B: one block per offset d; LDS-tiled separable 10x10 box-sum of the
// 8-channel pointwise product; emit masked distances, layout [q][d].
// ---------------------------------------------------------------------------
__global__ __launch_bounds__(256) void kdist(const float* __restrict__ xe,
                                             const float* __restrict__ ye,
                                             const float* __restrict__ n2q,
                                             const float* __restrict__ n2db,
                                             float* __restrict__ dist) {
  __shared__ float prod[XP * XSTR];
  __shared__ float rows[XP * HP];
  int d = blockIdx.x;
  int dy = d / 29 - WOFF, dx = d % 29 - WOFF;
  int tid = threadIdx.x;
  for (int i = tid; i < XP * XP; i += 256) {
    int r = i / XP, ci = i % XP;
    int yy = r - ADJ, xx = ci - ADJ;
    int y2 = yy + dy, x2 = xx + dx;
    float v = 0.f;
    if (yy >= 1 && yy <= HD && xx >= 1 && xx <= HD &&
        y2 >= 1 && y2 <= HD && x2 >= 1 && x2 <= HD) {
      int b1 = (yy - 1) * HD + (xx - 1);
      int b2 = (y2 - 1) * HD + (x2 - 1);
#pragma unroll
      for (int e = 0; e < NE; ++e) v += ye[e * HD * HD + b1] * xe[e * HD * HD + b2];
    }
    prod[r * XSTR + ci] = v;
  }
  __syncthreads();
  for (int i = tid; i < XP * HP; i += 256) {
    int r = i / HP, x = i % HP;
    float s = 0.f;
#pragma unroll
    for (int o = 0; o < PS; ++o) s += prod[r * XSTR + x + o];
    rows[r * HP + x] = s;
  }
  __syncthreads();
  bool selfo = (dy == 0 && dx == 0);
  for (int i = tid; i < NQ; i += 256) {
    int y = i / HP, x = i % HP;
    float cr = 0.f;
#pragma unroll
    for (int o = 0; o < PS; ++o) cr += rows[(y + o) * HP + x];
    int y2 = y + dy, x2 = x + dx;
    bool valid = (y2 >= 0 && y2 < HP && x2 >= 0 && x2 < HP);
    float n2s = valid ? n2db[y2 * HP + x2] : 0.f;
    float dv = (valid && !selfo) ? (n2q[i] + n2s - 2.f * cr) : 1e10f;
    dist[(size_t)i * NW + d] = dv;
  }
}

// ---------------------------------------------------------------------------
// Kernel C: per-query bitonic top-224 + 7 rounds stick-breaking softmax.
// ---------------------------------------------------------------------------
__global__ __launch_bounds__(256) void ksel(const float* __restrict__ dist,
                                            const float* __restrict__ ltv,
                                            int* __restrict__ idxq,
                                            float* __restrict__ Wq) {
  __shared__ float skey[1024];
  __shared__ int   sidx[1024];
  __shared__ float red[256];
  int q = blockIdx.x;
  int tid = threadIdx.x;
  const float* dq = dist + (size_t)q * NW;
  __builtin_prefetch(dq, 0, 3);
  for (int i = tid; i < 1024; i += 256) {
    skey[i] = (i < NW) ? dq[i] : 3.0e38f;
    sidx[i] = i;
  }
  for (int k = 2; k <= 1024; k <<= 1) {
    for (int s = k >> 1; s > 0; s >>= 1) {
      __syncthreads();
      for (int t = tid; t < 512; t += 256) {
        int i = (t / s) * (s * 2) + (t % s);
        int j = i + s;
        bool up = ((i & k) == 0);
        float ki = skey[i], kj = skey[j];
        int ii = sidx[i], ij = sidx[j];
        bool gt = (ki > kj) || (ki == kj && ii > ij);   // strict total order
        if (gt == up) { skey[i] = kj; skey[j] = ki; sidx[i] = ij; sidx[j] = ii; }
      }
    }
  }
  __syncthreads();
  float invT = __expf(-ltv[q]);           // logits = -dist / exp(lt)
  float lg = -3.0e38f;
  if (tid < KSEL) {
    lg = -skey[tid] * invT;
    idxq[(size_t)q * KSEL + tid] = sidx[tid];
  }
  for (int r = 0; r < NK; ++r) {
    red[tid] = (tid < KSEL) ? lg : -3.0e38f;
    __syncthreads();
    for (int s = 128; s > 0; s >>= 1) { if (tid < s) red[tid] = fmaxf(red[tid], red[tid + s]); __syncthreads(); }
    float mx = red[0];
    __syncthreads();
    red[tid] = (tid < KSEL) ? __expf(lg - mx) : 0.f;
    __syncthreads();
    for (int s = 128; s > 0; s >>= 1) { if (tid < s) red[tid] += red[tid + s]; __syncthreads(); }
    float sum = red[0];
    __syncthreads();
    if (tid < KSEL) {
      float w = __expf(lg - mx) / sum;
      Wq[((size_t)q * NK + r) * KSEL + tid] = w;
      lg += __logf(fmaxf(1.f - w, 1e-6f));
    }
  }
}

__global__ __launch_bounds__(256) void kzero(float* __restrict__ buf) {
  int i = blockIdx.x * blockDim.x + threadIdx.x;
  if (i < BUFN) buf[i] = 0.f;
}

// ---------------------------------------------------------------------------
// Kernel X: materialize xpad as f16 in global memory with zero borders and
// the LDS row stride (76) baked in, so the TDM can bulk-copy it 1:1 into LDS.
// ---------------------------------------------------------------------------
__global__ __launch_bounds__(256) void kxpad(const float* __restrict__ x,
                                             _Float16* __restrict__ xpadh) {
  int i = blockIdx.x * blockDim.x + threadIdx.x;
  if (i >= XPH) return;
  int cc = i / (XP * XSTR);
  int rr = (i / XSTR) % XP;
  int col = i % XSTR;
  float v = 0.f;
  if (col < XP && rr >= 6 && rr <= 69 && col >= 6 && col <= 69)
    v = x[(cc * HD + (rr - 6)) * HD + (col - 6)];
  xpadh[i] = (_Float16)v;
}

// TDM 1-D copy: nbytes (8-byte multiple) from global gaddr -> LDS ldsoff.
// D# fields per CDNA5 ISA: group0 {count=1, lds_addr,
// global_addr, type=2}; group1 {data_size=8B, tensor_dim0=tile_dim0=n/8,
// tensor_dim1=tile_dim1=1, dim0_stride=n/8}. Groups 2/3 zero (<=2D tensor).
__device__ __forceinline__ void tdm_load_1d(unsigned ldsoff,
                                            unsigned long long gaddr,
                                            unsigned nelem8) {
  v4u g0 = { 1u, ldsoff, (unsigned)gaddr,
             (unsigned)((gaddr >> 32) & 0x1FFFFFFull) | (2u << 30) };
  v8i g1 = { (int)(3u << 16),                 // data_size = 8 bytes
             (int)(nelem8 << 16),             // tensor_dim0[15:0] @ bits 63:48
             (int)(1u << 16),                 // tensor_dim1 = 1  @ bits 111:80
             (int)(nelem8 << 16),             // tile_dim0        @ bits 127:112
             1,                               // tile_dim1 = 1    @ bits 143:128
             (int)nelem8,                     // tensor_dim0_stride @ bits 207:160
             0, 0 };
  v4i gz = { 0, 0, 0, 0 };
#if __clang_major__ >= 23
  v8i gz8 = { 0, 0, 0, 0, 0, 0, 0, 0 };
  __builtin_amdgcn_tensor_load_to_lds(g0, g1, gz, gz, gz8, 0);
#else
  __builtin_amdgcn_tensor_load_to_lds(g0, g1, gz, gz, 0);
#endif
}

// ---------------------------------------------------------------------------
// Kernel E: WMMA aggregation. One wave per query: D(16x800) = A(16x224 f16,
// rows 7..15 zero) x B(224x800 f16) via v_wmma_f32_16x16x32_f16. B gathered
// from the query-independent xpad tile, TDM-staged in LDS as f16 (4 channels
// per pass). Per-neighbor LDS base offsets are precombined (u16) so each B
// element is one ds_load_u16 base + one ds_load_u16 data. Fragment index maps
// follow the CDNA5 ISA 16-bit A layout (05_wmma §7.12.2); B assumed to mirror
// A with N in place of M. Results scattered with global f32 atomics.
// ---------------------------------------------------------------------------
__global__ __launch_bounds__(256) void kagg(const _Float16* __restrict__ xpadh,
                                            const float* __restrict__ Wq,
                                            const int* __restrict__ idxq,
                                            float* __restrict__ buf) {
  __shared__ __align__(16) _Float16 xlds[XPASS];   // 45.6 KB: 4 channels
  __shared__ unsigned short nbase[8][KSEL];        // per-wave neighbor offsets
  int tid = threadIdx.x;
  int w = tid >> 5, lane = tid & 31;
  int q = blockIdx.x * 8 + w;
  bool active = q < NQ;
  int qy = active ? q / HP : 0;
  int qx = active ? q % HP : 0;

  if (active) {
    for (int j = lane; j < KSEL; j += 32) {
      int d = idxq[(size_t)q * KSEL + j];
      int ny = qy + d / 29 - WOFF;                 // [0,65]
      int nx = qx + d % 29 - WOFF;                 // [0,65]
      nbase[w][j] = (unsigned short)(ny * XSTR + nx);
    }
    __builtin_prefetch(&Wq[(size_t)q * NK * KSEL], 0, 3);
  }
  __syncthreads();

  // A fragments: 7 K-chunks of 32 neighbors, rows m>=7 zero-padded.
  v16h zero16 = {};
  v16h afr[NK];
  if (active) {
#pragma unroll
    for (int kc = 0; kc < NK; ++kc) {
      v16h a = {};
#pragma unroll
      for (int e = 0; e < 16; ++e) {
        int m  = lane & 15;
        int kk = ((lane >> 4) << 3) + (e & 7) + ((e >> 3) << 4);  // ISA A layout
        int j  = kc * 32 + kk;
        float v = (m < NK) ? Wq[((size_t)q * NK + m) * KSEL + j] : 0.f;
        a[e] = (_Float16)v;
      }
      afr[kc] = a;
    }
  } else {
#pragma unroll
    for (int kc = 0; kc < NK; ++kc) afr[kc] = zero16;
  }

  for (int p = 0; p < 2; ++p) {                 // channel passes: c in [4p,4p+4)
    __syncthreads();                            // pass p-1 LDS reads done
    if (w == 0) {                               // wave 0 issues the TDM copy
      tdm_load_1d((unsigned)(unsigned long long)&xlds[0],
                  (unsigned long long)(xpadh + p * XPASS),
                  (unsigned)(XPASS * sizeof(_Float16) / 8));   // 5700 x 8B
      __builtin_amdgcn_s_wait_tensorcnt(0);
    }
    __syncthreads();                            // LDS tile visible to all waves
    if (active) {
      for (int nt = 25 * p; nt < 25 * p + 25; ++nt) {
        int n   = nt * 16 + (lane & 15);        // this lane's output column
        int c   = n / 100;
        int rem = n % 100;
        int oy  = rem / 10, ox = rem % 10;
        int lanec = ((c - 4 * p) * XP + oy) * XSTR + ox;  // per-lane constant
        v8f acc = {};
#pragma unroll
        for (int kc = 0; kc < NK; ++kc) {
          v16h b = {};
#pragma unroll
          for (int e = 0; e < 16; ++e) {
            int kk = ((lane >> 4) << 3) + (e & 7) + ((e >> 3) << 4);
            int j  = kc * 32 + kk;
            b[e] = xlds[(int)nbase[w][j] + lanec];
          }
          acc = __builtin_amdgcn_wmma_f32_16x16x32_f16(
              false, afr[kc], false, b, (short)0, acc, false, false);
        }
        // Scatter accumulator: VGPR r, lane l -> row M = r + 8*(l>=16), col n.
#pragma unroll
        for (int r = 0; r < 8; ++r) {
          int M = r + ((lane >> 4) << 3);
          if (M < NK) {
            int bi = ((M * NC + c) * XP + (qy + oy)) * XP + (qx + ox);
            atomicAdd(&buf[bi], acc[r]);
          }
        }
      }
    }
  }
}

// ---------------------------------------------------------------------------
// Kernel F: normalize by analytic overlap count, subtract y, concat, crop.
// ---------------------------------------------------------------------------
__global__ __launch_bounds__(256) void kfin(const float* __restrict__ y,
                                            const float* __restrict__ buf,
                                            float* __restrict__ out) {
  int i = blockIdx.x * blockDim.x + threadIdx.x;
  if (i >= (1 + NK) * NC * HD * HD) return;
  int ch = i / (HD * HD);
  int rem = i % (HD * HD);
  int yy = rem / HD, xx = rem % HD;
  if (ch < NC) { out[i] = y[i]; return; }
  int m = ch - NC;                 // k*NC + c in 0..55
  int c = m % NC;
  int py = yy + 6, px = xx + 6;    // vid offset (crop 1 + ADJ 5)
  float ca = fminf((float)py, 9.f) - fmaxf(0.f, (float)(py - 65)) + 1.f;
  float cb = fminf((float)px, 9.f) - fmaxf(0.f, (float)(px - 65)) + 1.f;
  float v = buf[(m * XP + py) * XP + px] / (ca * cb);
  out[i] = v - y[(c * HD + yy) * HD + xx];
}

// ---------------------------------------------------------------------------
// Workspace layout (floats):
//   dist  : NQ*NW        = 3,663,396
//   n2q/n2db/ltv : 3*NQ  =    13,068
//   idxq  : NQ*KSEL ints =   975,744
//   Wq    : NQ*NK*KSEL   = 6,830,208
//   buf   : 56*75*75     =   315,000
//   xpadh : 8*75*76 f16  =    22,800 floats (8B-aligned offset)
// total ~47.3 MB.
// ---------------------------------------------------------------------------
extern "C" void kernel_launch(void* const* d_in, const int* in_sizes, int n_in,
                              void* d_out, int out_size, void* d_ws, size_t ws_size,
                              hipStream_t stream) {
  const float* x    = (const float*)d_in[0];
  const float* xe   = (const float*)d_in[1];
  const float* ye   = (const float*)d_in[2];
  const float* y    = (const float*)d_in[3];
  const float* ltin = (const float*)d_in[4];

  float* ws    = (float*)d_ws;
  float* dist  = ws;
  float* n2q   = dist + (size_t)NQ * NW;
  float* n2db  = n2q + NQ;
  float* ltv   = n2db + NQ;
  int*   idxq  = (int*)(ltv + NQ);
  float* Wq    = (float*)(idxq + (size_t)NQ * KSEL);
  float* buf   = Wq + (size_t)NQ * NK * KSEL;
  _Float16* xpadh = (_Float16*)(buf + BUFN);

  kprep<<<(NQ + 255) / 256, 256, 0, stream>>>(xe, ye, ltin, n2q, n2db, ltv);
  kdist<<<NW, 256, 0, stream>>>(xe, ye, n2q, n2db, dist);
  ksel<<<NQ, 256, 0, stream>>>(dist, ltv, idxq, Wq);
  kzero<<<(BUFN + 255) / 256, 256, 0, stream>>>(buf);
  kxpad<<<(XPH + 255) / 256, 256, 0, stream>>>(x, xpadh);
  kagg<<<(NQ + 7) / 8, 256, 0, stream>>>(xpadh, Wq, idxq, buf);
  kfin<<<((1 + NK) * NC * HD * HD + 255) / 256, 256, 0, stream>>>(y, buf, (float*)d_out);
}
